// HMM_EM_75067438399546
// MI455X (gfx1250) — compile-verified, matrix operand order checked by reading, not run on hardware
//
#include <hip/hip_runtime.h>

#define ZD    512
#define XD    10000
#define SEQ   256
#define BATCH 64
#define RS    520      // P row stride in bf16 elems (1040B: +16B pad -> 4-bank rotation per row)
#define NWAVE 8

typedef __attribute__((ext_vector_type(16))) __bf16 v16bf;
typedef __attribute__((ext_vector_type(8)))  float  v8f;

static __device__ __forceinline__ unsigned short f2bf(float f) {
  union { float f; unsigned u; } c; c.f = f;
  unsigned u = c.u;
  u += 0x7FFFu + ((u >> 16) & 1u);   // round-to-nearest-even
  return (unsigned short)(u >> 16);
}

static __device__ __forceinline__ v16bf mkv(uint4 a, uint4 b) {
  union { uint4 q[2]; v16bf v; } u;
  u.q[0] = a; u.q[1] = b;
  return u.v;
}

// ---------------------------------------------------------------------------
// Kernel 1: colLse[i] = logsumexp_j T[j,i]   (log_softmax(T, axis=0) denominator)
// ---------------------------------------------------------------------------
__global__ void colLse_kernel(const float* __restrict__ T, float* __restrict__ colLse) {
  int i = blockIdx.x;          // column
  int tid = threadIdx.x;       // 128 threads
  __shared__ float red[128];
  float m = -__builtin_inff();
  for (int j = tid; j < ZD; j += 128) m = fmaxf(m, T[(size_t)j * ZD + i]);
  red[tid] = m; __syncthreads();
  for (int s = 64; s > 0; s >>= 1) { if (tid < s) red[tid] = fmaxf(red[tid], red[tid + s]); __syncthreads(); }
  m = red[0]; __syncthreads();
  float ssum = 0.f;
  for (int j = tid; j < ZD; j += 128) ssum += expf(T[(size_t)j * ZD + i] - m);
  red[tid] = ssum; __syncthreads();
  for (int s = 64; s > 0; s >>= 1) { if (tid < s) red[tid] += red[tid + s]; __syncthreads(); }
  if (tid == 0) colLse[i] = m + logf(red[0]);
}

// ---------------------------------------------------------------------------
// Kernel 2: pack M[i,j] = exp(T[j,i] - colLse[i]) as bf16 in WMMA B-operand
// tile layout: tile (jt,ic) = 32x16 (KxN) chunk; per lane (n=l&15,h=l>>4) the
// 8 dwords are the exact VGPR order of the 16-bit B operand, so the consumer
// loads each B tile as two contiguous b128s.
// ---------------------------------------------------------------------------
__global__ void packM_kernel(const float* __restrict__ T, const float* __restrict__ colLse,
                             unsigned short* __restrict__ Mt) {
  int tile = blockIdx.x;            // jt*16 + ic, 512 tiles
  int jt = tile >> 4, ic = tile & 15;
  int lane = threadIdx.x;           // 32
  int n = lane & 15, h = lane >> 4;
  int j  = jt * 16 + n;
  int i0 = ic * 32;
  const float* Trow = T + (size_t)j * ZD + i0;   // T[j][i0..i0+31], contiguous
  unsigned pw[8];
#pragma unroll
  for (int d = 0; d < 8; ++d) {
    int base = ((d & 4) ? 16 : 0) + 8 * h + 2 * (d & 3);  // K index of the dword pair
    float v0 = expf(Trow[base]     - colLse[i0 + base]);
    float v1 = expf(Trow[base + 1] - colLse[i0 + base + 1]);
    pw[d] = (unsigned)f2bf(v0) | ((unsigned)f2bf(v1) << 16);
  }
  uint4* dst = (uint4*)(Mt + ((size_t)tile * 32 + lane) * 16);
  dst[0] = make_uint4(pw[0], pw[1], pw[2], pw[3]);
  dst[1] = make_uint4(pw[4], pw[5], pw[6], pw[7]);
}

// ---------------------------------------------------------------------------
// Kernel 3: emitLse[z] = logsumexp_x emit[x,z]  (online logsumexp per thread)
// ---------------------------------------------------------------------------
__global__ void emitLse_kernel(const float* __restrict__ emit, float* __restrict__ emitLse) {
  int z = blockIdx.x;
  int tid = threadIdx.x;   // 256
  float m = -__builtin_inff(), s = 0.f;
  for (int x = tid; x < XD; x += 256) {
    float v = emit[(size_t)x * ZD + z];
    if (v > m) { s = s * __expf(m - v) + 1.f; m = v; }
    else       { s += __expf(v - m); }
  }
  __shared__ float rm[256], rs[256];
  rm[tid] = m; rs[tid] = s; __syncthreads();
  for (int st = 128; st > 0; st >>= 1) {
    if (tid < st) {
      float m2 = rm[tid + st], s2 = rs[tid + st];
      float M = fmaxf(rm[tid], m2);
      rs[tid] = rs[tid] * __expf(rm[tid] - M) + s2 * __expf(m2 - M);
      rm[tid] = M;
    }
    __syncthreads();
  }
  if (tid == 0) emitLse[z] = rm[0] + logf(rs[0]);
}

// ---------------------------------------------------------------------------
// Kernel 4: persistent forward recurrence. One workgroup per 16-row batch
// slice (recurrence is independent across batch). 8 waves; wave w owns
// j-tiles 4w..4w+3. P = exp(alpha - rowmax) kept in LDS as bf16 (A operand),
// M streamed from L2 (B operand). 64 v_wmma_f32_16x16x32_bf16 per wave/step.
// ---------------------------------------------------------------------------
__global__ __launch_bounds__(256, 1)
void hmm_fwd_kernel(const int* __restrict__ ids, const float* __restrict__ pi,
                    const float* __restrict__ emit, const unsigned short* __restrict__ Mt,
                    const float* __restrict__ emitLse, float* __restrict__ out)
{
  __shared__ __align__(16) unsigned short P[16 * RS];  // 16.6 KB
  __shared__ float rowpart[16][9];                     // per-wave row maxes, col 8 = mu
  __shared__ float CL[16];                             // running log-scale per batch row
  __shared__ float red[NWAVE];
  __shared__ float sval;

  const int tid   = threadIdx.x;
  const int lane  = tid & 31;
  const int w     = tid >> 5;       // wave 0..7
  const int n     = lane & 15;      // N / row-selector within tile
  const int h     = lane >> 4;      // half
  const int bbase = blockIdx.x * 16;

  // ---- log-softmax(pi) denominator (block-wide logsumexp over 512) ----
  float v1 = pi[tid], v2 = pi[tid + 256];
  float m = fmaxf(v1, v2);
#pragma unroll
  for (int off = 16; off; off >>= 1) m = fmaxf(m, __shfl_xor(m, off, 32));
  if (lane == 0) red[w] = m;
  __syncthreads();
  if (tid == 0) { float mm = red[0]; for (int k = 1; k < NWAVE; ++k) mm = fmaxf(mm, red[k]); sval = mm; }
  __syncthreads();
  const float pim = sval;
  float es = __expf(v1 - pim) + __expf(v2 - pim);
#pragma unroll
  for (int off = 16; off; off >>= 1) es += __shfl_xor(es, off, 32);
  if (lane == 0) red[w] = es;
  if (tid < 16) CL[tid] = 0.f;
  __syncthreads();
  if (tid == 0) { float ss = 0.f; for (int k = 0; k < NWAVE; ++k) ss += red[k]; sval = pim + logf(ss); }
  __syncthreads();
  const float pilse = sval;

  int   colq[4];
  float lseJ[4], piL[4];
#pragma unroll
  for (int q = 0; q < 4; ++q) {
    colq[q] = (w * 4 + q) * 16 + n;
    lseJ[q] = emitLse[colq[q]];
    piL[q]  = pi[colq[q]] - pilse;
  }

  float raw[4][8];   // this thread's slots of raw[b,j] = obs + log S (alpha - C)

  for (int t = 0; t < SEQ; ++t) {
    // -- issue the obs gather first so it overlaps the GEMM --
    int myid[8];
#pragma unroll
    for (int v = 0; v < 8; ++v) myid[v] = ids[t * BATCH + bbase + v + 8 * h];
    float obs[4][8];
#pragma unroll
    for (int q = 0; q < 4; ++q)
#pragma unroll
      for (int v = 0; v < 8; ++v)
        obs[q][v] = emit[(size_t)myid[v] * ZD + colq[q]];

    if (t == 0) {
#pragma unroll
      for (int q = 0; q < 4; ++q)
#pragma unroll
        for (int v = 0; v < 8; ++v)
          raw[q][v] = obs[q][v] - lseJ[q] + piL[q];
    } else {
      v8f zz = {0.f, 0.f, 0.f, 0.f, 0.f, 0.f, 0.f, 0.f};
      v8f acc[4] = {zz, zz, zz, zz};
      const char* Pch = (const char*)P + (size_t)n * (RS * 2);
#pragma unroll
      for (int kc = 0; kc < 16; ++kc) {
        // A: P[16 x 32] bf16, per-lane VGPR layout = two 16B LDS chunks
        uint4 a0 = *(const uint4*)(Pch + kc * 64 + h * 16);
        uint4 a1 = *(const uint4*)(Pch + kc * 64 + 32 + h * 16);
        v16bf Am = mkv(a0, a1);
#pragma unroll
        for (int q = 0; q < 4; ++q) {
          const uint4* bp = (const uint4*)(Mt + (((size_t)(w * 4 + q) * 16 + kc) * 32 + lane) * 16);
          v16bf Bm = mkv(bp[0], bp[1]);
          acc[q] = __builtin_amdgcn_wmma_f32_16x16x32_bf16(
              false, Am, false, Bm, (short)0, acc[q], false, false);
        }
        if (kc < 15)
          __builtin_prefetch((const void*)(Mt + (((size_t)(w * 4) * 16 + kc + 1) * 32 + lane) * 16), 0, 0);
      }
#pragma unroll
      for (int q = 0; q < 4; ++q)
#pragma unroll
        for (int v = 0; v < 8; ++v)
          raw[q][v] = obs[q][v] - lseJ[q] + __logf(acc[q][v]);
    }

    // -- per-row max over all 512 j: in-thread (4 tiles) -> half-wave -> waves --
#pragma unroll
    for (int v = 0; v < 8; ++v) {
      float mr = fmaxf(fmaxf(raw[0][v], raw[1][v]), fmaxf(raw[2][v], raw[3][v]));
#pragma unroll
      for (int off = 1; off < 16; off <<= 1) mr = fmaxf(mr, __shfl_xor(mr, off, 32));
      if (n == 0) rowpart[v + 8 * h][w] = mr;
    }
    __syncthreads();
    if (tid < 16) {
      float mm = rowpart[tid][0];
#pragma unroll
      for (int k = 1; k < NWAVE; ++k) mm = fmaxf(mm, rowpart[tid][k]);
      rowpart[tid][8] = mm;       // mu for this row
      CL[tid] += mm;              // accumulate log-scale
    }
    __syncthreads();

    // -- write normalized probabilities P = exp(raw - mu) for next step --
#pragma unroll
    for (int v = 0; v < 8; ++v) {
      float mu = rowpart[v + 8 * h][8];
#pragma unroll
      for (int q = 0; q < 4; ++q)
        P[(v + 8 * h) * RS + colq[q]] = f2bf(__expf(raw[q][v] - mu));
    }
    __syncthreads();
  }

  // ---- finalize: -log p = -(C + log sum_j exp(raw_last - mu_last)) ----
#pragma unroll
  for (int v = 0; v < 8; ++v) {
    float mu = rowpart[v + 8 * h][8];
    float se = 0.f;
#pragma unroll
    for (int q = 0; q < 4; ++q) se += __expf(raw[q][v] - mu);
#pragma unroll
    for (int off = 1; off < 16; off <<= 1) se += __shfl_xor(se, off, 32);
    if (n == 0) rowpart[v + 8 * h][w] = se;
  }
  __syncthreads();
  if (tid < 16) {
    float s = 0.f;
#pragma unroll
    for (int k = 0; k < NWAVE; ++k) s += rowpart[tid][k];
    out[bbase + tid] = -(CL[tid] + logf(s));
  }
}

// ---------------------------------------------------------------------------
extern "C" void kernel_launch(void* const* d_in, const int* in_sizes, int n_in,
                              void* d_out, int out_size, void* d_ws, size_t ws_size,
                              hipStream_t stream) {
  const int*   ids  = (const int*)d_in[0];     // [256,64] int32
  const float* T    = (const float*)d_in[1];   // [512,512]
  const float* pi   = (const float*)d_in[2];   // [512]
  const float* emit = (const float*)d_in[3];   // [10000,512]
  float* out = (float*)d_out;                  // [64]

  // workspace: Mt (bf16 packed, 512 KB) | colLse (2 KB) | emitLse (2 KB)
  unsigned short* Mt      = (unsigned short*)d_ws;
  float*          colLse  = (float*)((char*)d_ws + 512 * 1024);
  float*          emitLse = colLse + ZD;

  colLse_kernel <<<ZD, 128, 0, stream>>>(T, colLse);
  packM_kernel  <<<ZD, 32,  0, stream>>>(T, colLse, Mt);
  emitLse_kernel<<<ZD, 256, 0, stream>>>(emit, emitLse);
  hmm_fwd_kernel<<<BATCH / 16, 256, 0, stream>>>(ids, pi, emit, Mt, emitLse, out);
}
